// Quantizer_20169166422747
// MI455X (gfx1250) — matrix operand order repaired and checked by reading, not compiled, not run on hardware
//
#include <hip/hip_runtime.h>
#include <hip/hip_bf16.h>
#include <stdint.h>

// Problem constants (from reference): B=16, T=4096, D=64, K=1024
#define D_ 64
#define K_ 1024
#define N_ 65536            // B*T
#define TILES_ (K_ / 16)    // 64 code tiles of 16 columns

// LDS layout for vq_argmax (dynamic shared segment):
// codebook rows padded to 72 bf16 (144B) -> conflict-free B-frag reads
#define ROWB_   144                      // padded row pitch in bytes
#define HIOFF_  0                        // embHi: 1024 * 144 = 147456 B
#define LOOFF_  147456                   // embLo: 147456 B
#define NHOFF_  294912                   // negHalf: 4096 B
#define SMEMB_  299008                   // total dynamic LDS bytes (~292 KB < 320 KB)

typedef __attribute__((ext_vector_type(16))) __bf16 v16bf;
typedef __attribute__((ext_vector_type(8)))  float  v8f;

// gfx1250 async global->LDS DMA (tracked by ASYNCcnt). Inline asm keeps it
// portable across ROCm 7.2 / amdgpu-toolchain builtin differences.
__device__ __forceinline__ void async_copy_b128(uint32_t ldsOff, const void* g) {
  asm volatile("global_load_async_to_lds_b128 %0, %1, off"
               :: "v"(ldsOff), "v"((uint64_t)(uintptr_t)g)
               : "memory");
}
__device__ __forceinline__ void wait_asynccnt0() {
  asm volatile("s_wait_asynccnt 0x0" ::: "memory");
}

// ---------------------------------------------------------------------------
// Kernel 1: codebook prep. One wave32 per code row:
//   emb(f32) -> embHi/embLo (bf16 hi/lo split), negHalf[k] = -0.5*||e_k||^2
// ---------------------------------------------------------------------------
__global__ void __launch_bounds__(256) vq_prep(const float* __restrict__ emb,
                                               __bf16* __restrict__ embHi,
                                               __bf16* __restrict__ embLo,
                                               float* __restrict__ negHalf) {
  const int wave = (blockIdx.x * blockDim.x + threadIdx.x) >> 5;  // 0..1023
  const int lane = threadIdx.x & 31;
  const float* row = emb + (size_t)wave * D_;
  float2 v = *(const float2*)(row + lane * 2);
  __bf16 h0 = (__bf16)v.x;
  __bf16 h1 = (__bf16)v.y;
  __bf16 l0 = (__bf16)(v.x - (float)h0);
  __bf16 l1 = (__bf16)(v.y - (float)h1);
  size_t o = (size_t)wave * D_ + lane * 2;
  embHi[o]     = h0;  embHi[o + 1] = h1;
  embLo[o]     = l0;  embLo[o + 1] = l1;
  float s = v.x * v.x + v.y * v.y;
  #pragma unroll
  for (int off = 16; off > 0; off >>= 1) s += __shfl_xor(s, off, 32);
  if (lane == 0) negHalf[wave] = -0.5f * s;
}

// ---------------------------------------------------------------------------
// Kernel 2: argmax of (z.e - 0.5||e||^2) via bf16 hi/lo-split WMMA (fp32 acc).
// 512 threads = 16 waves; whole hi/lo codebook staged into LDS once per block
// via async global->LDS DMA, then each wave owns 16 rows of z and sweeps all
// 64 code tiles, reading B fragments from LDS (bank-conflict-free padding).
// Software-pipelined: tile t+1's B fragments are loaded from LDS while tile
// t's WMMAs execute, so ds waits overlap the XDL chain.
// A-frag (16-bit A 16x32): lane = row (lane&15); half-wave K-chunks of 16.
// B-frag (16-bit B 32x16): lane = col (lane&15); half-wave K-chunks of 16.
// ---------------------------------------------------------------------------
__global__ void __launch_bounds__(512) vq_argmax(const float* __restrict__ z,
                                                 const __bf16* __restrict__ embHi,
                                                 const __bf16* __restrict__ embLo,
                                                 const float* __restrict__ negHalf,
                                                 int* __restrict__ bestIdxOut) {
  extern __shared__ __align__(16) char smem[];
  const int tid  = threadIdx.x;
  const int lane = tid & 31;
  const int wave = tid >> 5;                          // 0..15
  const int rowBlock = blockIdx.x * 256 + wave * 16;  // 16 rows per wave
  const int m    = lane & 15;
  const int half = lane >> 4;

  // ---- Stage codebook into LDS (async DMA, ASYNCcnt) -----------------------
  // 8192 16-byte chunks per table; 16 chunks per thread per table.
  const uint32_t ldsBase = __builtin_amdgcn_groupstaticsize();  // dyn seg base (0)
  #pragma unroll 4
  for (int it = 0; it < 16; ++it) {
    int c = it * 512 + tid;
    int row = c >> 3, col = c & 7;
    async_copy_b128(ldsBase + HIOFF_ + row * ROWB_ + col * 16,
                    embHi + (size_t)row * D_ + col * 8);
  }
  #pragma unroll 4
  for (int it = 0; it < 16; ++it) {
    int c = it * 512 + tid;
    int row = c >> 3, col = c & 7;
    async_copy_b128(ldsBase + LOOFF_ + row * ROWB_ + col * 16,
                    embLo + (size_t)row * D_ + col * 8);
  }

  // ---- Build persistent A fragments while the async DMA is in flight -------
  // K-tiles: kt=0 covers D 0..31, kt=1 covers D 32..63.
  const float* zr = z + (size_t)(rowBlock + m) * D_;
  const int kb = half * 16;
  v16bf aHi[2], aLo[2];
  #pragma unroll
  for (int kt = 0; kt < 2; ++kt) {
    #pragma unroll
    for (int c = 0; c < 2; ++c) {                     // chunk: d = kt*32 + kb + c*8
      const float* p = zr + kt * 32 + kb + c * 8;
      #pragma unroll
      for (int j = 0; j < 8; ++j) {
        float f  = p[j];
        __bf16 h = (__bf16)f;
        aHi[kt][c * 8 + j] = h;
        aLo[kt][c * 8 + j] = (__bf16)(f - (float)h);
      }
    }
  }

  #pragma unroll
  for (int i = tid; i < K_; i += 512)                 // bias table (plain ds_store)
    *(float*)(smem + NHOFF_ + i * 4) = negHalf[i];
  wait_asynccnt0();
  __syncthreads();

  float best[8];
  int   bidx[8];
  #pragma unroll
  for (int i = 0; i < 8; ++i) { best[i] = -3.4e38f; bidx[i] = 0; }

  const uint32_t kOffB = (uint32_t)half * 32;         // B half-wave K offset (bytes)

  // Prologue: load tile 0's B fragments + bias.
  v16bf bHi[2], bLo[2];
  {
    const uint32_t b0 = (uint32_t)m * ROWB_ + kOffB;
    #pragma unroll
    for (int kt = 0; kt < 2; ++kt) {
      bHi[kt] = *(const v16bf*)(smem + HIOFF_ + b0 + kt * 64);
      bLo[kt] = *(const v16bf*)(smem + LOOFF_ + b0 + kt * 64);
    }
  }
  float bias = *(const float*)(smem + NHOFF_ + m * 4);

  for (int t = 0; t < TILES_; ++t) {
    // -- Prefetch tile t+1 (wrapped, branch-free: EXEC stays all-ones) -------
    const int tn = (t + 1) & (TILES_ - 1);
    const int ncode = tn * 16 + m;
    const uint32_t nb = (uint32_t)ncode * ROWB_ + kOffB;
    v16bf nHi[2], nLo[2];
    #pragma unroll
    for (int kt = 0; kt < 2; ++kt) {
      nHi[kt] = *(const v16bf*)(smem + HIOFF_ + nb + kt * 64);
      nLo[kt] = *(const v16bf*)(smem + LOOFF_ + nb + kt * 64);
    }
    const float nbias = *(const float*)(smem + NHOFF_ + ncode * 4);

    // -- Compute tile t with resident fragments ------------------------------
    v8f c1 = { 0.f, 0.f, 0.f, 0.f, 0.f, 0.f, 0.f, 0.f };  // hi*hi chain
    v8f c2 = { 0.f, 0.f, 0.f, 0.f, 0.f, 0.f, 0.f, 0.f };  // cross-term chain
    #pragma unroll
    for (int kt = 0; kt < 2; ++kt) {
      c1 = __builtin_amdgcn_wmma_f32_16x16x32_bf16(false, aHi[kt], false, bHi[kt],
                                                   (short)0, c1, false, false);
      c2 = __builtin_amdgcn_wmma_f32_16x16x32_bf16(false, aHi[kt], false, bLo[kt],
                                                   (short)0, c2, false, false);
      c2 = __builtin_amdgcn_wmma_f32_16x16x32_bf16(false, aLo[kt], false, bHi[kt],
                                                   (short)0, c2, false, false);
    }
    const int code = t * 16 + m;
    #pragma unroll
    for (int i = 0; i < 8; ++i) {
      float s  = (c1[i] + c2[i]) + bias;
      bool  gt = s > best[i];                 // strict > keeps earliest (smallest) index
      best[i] = gt ? s : best[i];
      bidx[i] = gt ? code : bidx[i];
    }

    // -- Rotate double buffers ----------------------------------------------
    #pragma unroll
    for (int kt = 0; kt < 2; ++kt) { bHi[kt] = nHi[kt]; bLo[kt] = nLo[kt]; }
    bias = nbias;
  }

  // Reduce max/argmax across the 16 lanes holding a row's columns.
  // C/D layout: VGPR i -> row i (lanes 0-15) / row i+8 (lanes 16-31).
  #pragma unroll
  for (int i = 0; i < 8; ++i) {
    float v  = best[i];
    int   id = bidx[i];
    #pragma unroll
    for (int off = 1; off < 16; off <<= 1) {
      float v2 = __shfl_xor(v, off, 32);
      int   i2 = __shfl_xor(id, off, 32);
      bool take = (v2 > v) || (v2 == v && i2 < id);  // argmin-style tiebreak
      v  = take ? v2 : v;
      id = take ? i2 : id;
    }
    if (m == 0) bestIdxOut[rowBlock + half * 8 + i] = id;
  }
}

// ---------------------------------------------------------------------------
// Kernel 3: gather quantized vectors (fp32, exact), write indices as float,
// deterministic per-block partial sums of masked SSE and mask count.
// 4 threads per row, 16 floats each; 64 rows per 256-thread block.
// ---------------------------------------------------------------------------
__global__ void __launch_bounds__(256) vq_gather(const float* __restrict__ z,
                                                 const unsigned char* __restrict__ mask,
                                                 const float* __restrict__ emb,
                                                 const int* __restrict__ bestIdx,
                                                 float* __restrict__ outQ,
                                                 float* __restrict__ outIdx,
                                                 float* __restrict__ psse,
                                                 float* __restrict__ pcnt) {
  __shared__ float ssum[256];
  __shared__ float scnt[256];
  const int tid  = threadIdx.x;
  const int g    = blockIdx.x * 256 + tid;
  const int n    = g >> 2;
  const int part = g & 3;
  const int d0   = part * 16;

  const int   idx  = bestIdx[n];
  const float mval = mask[n] ? 1.0f : 0.0f;
  const float* e  = emb + (size_t)idx * D_ + d0;
  const float* zp = z   + (size_t)n   * D_ + d0;
  float*       op = outQ + (size_t)n  * D_ + d0;

  float acc = 0.f;
  #pragma unroll
  for (int j = 0; j < 4; ++j) {
    float4 ev = ((const float4*)e)[j];
    float4 zv = ((const float4*)zp)[j];
    ((float4*)op)[j] = ev;
    float dx = ev.x - zv.x, dy = ev.y - zv.y, dz = ev.z - zv.z, dw = ev.w - zv.w;
    acc += dx * dx + dy * dy + dz * dz + dw * dw;
  }
  ssum[tid] = acc * mval;
  scnt[tid] = (part == 0) ? mval : 0.f;
  if (part == 0) outIdx[n] = (float)idx;
  __syncthreads();
  #pragma unroll
  for (int s = 128; s > 0; s >>= 1) {
    if (tid < s) { ssum[tid] += ssum[tid + s]; scnt[tid] += scnt[tid + s]; }
    __syncthreads();
  }
  if (tid == 0) { psse[blockIdx.x] = ssum[0]; pcnt[blockIdx.x] = scnt[0]; }
}

// ---------------------------------------------------------------------------
// Kernel 4: final deterministic reduction -> loss = 1.25 * sse / (cnt * 64)
// ---------------------------------------------------------------------------
__global__ void __launch_bounds__(256) vq_final(const float* __restrict__ psse,
                                                const float* __restrict__ pcnt,
                                                int nb, float* __restrict__ lossOut) {
  __shared__ float ss[256];
  __shared__ float sc[256];
  float a = 0.f, c = 0.f;
  for (int i = threadIdx.x; i < nb; i += 256) { a += psse[i]; c += pcnt[i]; }
  ss[threadIdx.x] = a;
  sc[threadIdx.x] = c;
  __syncthreads();
  #pragma unroll
  for (int s = 128; s > 0; s >>= 1) {
    if (threadIdx.x < s) { ss[threadIdx.x] += ss[threadIdx.x + s];
                           sc[threadIdx.x] += sc[threadIdx.x + s]; }
    __syncthreads();
  }
  if (threadIdx.x == 0) {
    float denom = sc[0] * (float)D_;
    lossOut[0] = (denom > 0.f) ? (1.25f * ss[0] / denom) : 0.f;
  }
}

// ---------------------------------------------------------------------------
extern "C" void kernel_launch(void* const* d_in, const int* in_sizes, int n_in,
                              void* d_out, int out_size, void* d_ws, size_t ws_size,
                              hipStream_t stream) {
  (void)in_sizes; (void)n_in; (void)out_size; (void)ws_size;
  const float*         z    = (const float*)d_in[0];          // [B,T,D] f32
  const unsigned char* mask = (const unsigned char*)d_in[1];  // [B,T] bool (1 byte)
  const float*         emb  = (const float*)d_in[2];          // [K,D] f32

  // Workspace layout (bytes):
  //   [0,128K)      embHi  bf16[K][D]
  //   [128K,256K)   embLo  bf16[K][D]
  //   [256K,260K)   negHalf f32[K]
  //   [272K,528K)   bestIdx int[N]
  //   [528K,532K)   psse f32[1024]
  //   [532K,536K)   pcnt f32[1024]
  char*    w       = (char*)d_ws;
  __bf16*  embHi   = (__bf16*)w;
  __bf16*  embLo   = embHi + (size_t)K_ * D_;
  float*   negHalf = (float*)(w + 256 * 1024);
  int*     bestIdx = (int*)(w + 272 * 1024);
  float*   psse    = (float*)(w + 528 * 1024);
  float*   pcnt    = (float*)(w + 532 * 1024);

  float* outQ    = (float*)d_out;                 // [N, D]
  float* outIdx  = outQ + (size_t)N_ * D_;        // [N] indices as float
  float* lossOut = outIdx + N_;                   // [1]

  vq_prep  <<<K_ / 8,       256, 0,      stream>>>(emb, embHi, embLo, negHalf);
  vq_argmax<<<N_ / 256,     512, SMEMB_, stream>>>(z, embHi, embLo, negHalf, bestIdx);
  vq_gather<<<N_ * 4 / 256, 256, 0,      stream>>>(z, mask, emb, bestIdx, outQ, outIdx,
                                                   psse, pcnt);
  vq_final <<<1,            256, 0,      stream>>>(psse, pcnt, 1024, lossOut);
}
